// AttentionPaged_26242250179100
// MI455X (gfx1250) — matrix operand order repaired and checked by reading, not compiled, not run on hardware
//
#include <hip/hip_runtime.h>
#include <hip/hip_bf16.h>
#include <cstdint>

// ---------------- problem constants ----------------
constexpr int Bsz   = 8;
constexpr int S     = 1024;
constexpr int T     = Bsz * S;          // 8192 tokens
constexpr int HID   = 4096;
constexpr int NH    = 32;
constexpr int NKV   = 8;
constexpr int HD    = 128;
constexpr int G     = NH / NKV;         // 4
constexpr int QKVD  = (NH + 2 * NKV) * HD; // 6144
constexpr int CACHE_ROWS = 1024 * 16;   // NUM_BLOCKS*BLOCK_SIZE = 16384
constexpr float ROPE_THETA = 10000.0f;

// ---------------- fragment types (wave32 WMMA) ----------------
typedef __attribute__((ext_vector_type(16))) __bf16 bf16x16;
typedef __attribute__((ext_vector_type(8)))  float  f32x8;
typedef __attribute__((ext_vector_type(4)))  unsigned int u32x4;

union Frag16 {              // 16 bf16 = one A or B operand of 16x16x32 bf16 WMMA
  bf16x16 v;
  u32x4   q[2];
  __bf16  h[16];
};

__device__ __forceinline__ f32x8 wmma_bf16(const Frag16& a, const Frag16& b, f32x8 c) {
  return __builtin_amdgcn_wmma_f32_16x16x32_bf16(
      /*neg_a=*/false, a.v, /*neg_b=*/false, b.v,
      /*c_mod=*/(short)0, c, /*reuse_a=*/false, /*reuse_b=*/false);
}

// ---------------- gfx1250 async copy global -> LDS (ASYNCcnt tracked) ----------
// Generic pointers to __shared__ carry the wave-relative LDS byte offset in their
// low 32 bits (ISA 10.2); that is what VDST must hold.
__device__ __forceinline__ unsigned lds_off32(const void* p) {
  return (unsigned)(uintptr_t)p;
}
__device__ __forceinline__ void async_load_lds_b128(unsigned ldsoff, const void* gptr) {
  asm volatile("global_load_async_to_lds_b128 %0, %1, off"
               :: "v"(ldsoff), "v"(gptr)
               : "memory");
}
__device__ __forceinline__ void wait_asynccnt0() {
  asm volatile("s_wait_asynccnt 0x0" ::: "memory");
}

// ---------------- utility kernels ----------------
__global__ void f32_to_bf16_kernel(const float* __restrict__ src,
                                   __bf16* __restrict__ dst, size_t n) {
  size_t i = (size_t)blockIdx.x * blockDim.x + threadIdx.x;
  size_t stride = (size_t)gridDim.x * blockDim.x;
  for (; i < n; i += stride) dst[i] = (__bf16)src[i];
}

__global__ void zero_f32_kernel(float* __restrict__ dst, size_t n) {
  size_t i = (size_t)blockIdx.x * blockDim.x + threadIdx.x;
  size_t stride = (size_t)gridDim.x * blockDim.x;
  for (; i < n; i += stride) dst[i] = 0.f;
}

// ---------------- bf16 WMMA GEMM: C[M,N] = A[M,K] @ B[K,N] (+bias) ----------------
// block: 256 threads = 8 waves in a 2(M) x 4(N) grid; block tile 128x128, K-chunk 32.
// wave tile: 64(M) x 32(N) = 4x2 WMMA subtiles. A tile staged with async-to-LDS.
__global__ __launch_bounds__(256) void gemm_bf16_wmma(
    const __bf16* __restrict__ A, const __bf16* __restrict__ B,
    const float* __restrict__ bias, float* __restrict__ C,
    int M, int N, int K) {
  constexpr int LDT = 40; // bf16 elements per LDS row (32 + pad, keeps 16B alignment)
  __shared__ __bf16 As[128 * LDT];       // [row][k], K contiguous
  __shared__ __bf16 Bs[128 * LDT];       // transposed: [n][k], K contiguous

  const int tid  = threadIdx.x;
  const int lane = tid & 31;
  const int w    = tid >> 5;
  const int wm   = w >> 2, wn = w & 3;
  const int g    = lane >> 4, r = lane & 15;
  const int m0   = blockIdx.y * 128, n0 = blockIdx.x * 128;

  f32x8 acc[4][2];
  for (int i = 0; i < 4; i++)
    for (int j = 0; j < 2; j++)
      for (int e = 0; e < 8; e++) acc[i][j][e] = 0.f;

  const int arow  = tid >> 1, akoff = (tid & 1) * 16;  // A: 128 rows x 32 k
  const int bkrow = tid >> 3, bnoff = (tid & 7) * 16;  // B: 32 k-rows x 128 n
  const unsigned lds_a = lds_off32(&As[arow * LDT + akoff]);

  for (int k0 = 0; k0 < K; k0 += 32) {
    if (k0 + 32 < K) {  // gfx1250 global_prefetch_b8 for the next tiles
      __builtin_prefetch(A + (size_t)(m0 + arow) * K + k0 + 32 + akoff, 0, 1);
      __builtin_prefetch(B + (size_t)(k0 + 32 + bkrow) * N + n0 + bnoff, 0, 1);
    }
    {  // stage A tile: async global -> LDS, 2 x b128 per lane
      const __bf16* ga = A + (size_t)(m0 + arow) * K + k0 + akoff;
      async_load_lds_b128(lds_a, ga);
      async_load_lds_b128(lds_a + 16, ga + 8);
    }
    {  // stage B tile transposed -> [n][k] (register path, transpose on store)
      Frag16 t;
      const u32x4* src = (const u32x4*)(B + (size_t)(k0 + bkrow) * N + n0 + bnoff);
      t.q[0] = src[0];
      t.q[1] = src[1];
#pragma unroll
      for (int e = 0; e < 16; e++) Bs[(bnoff + e) * LDT + bkrow] = t.h[e];
    }
    wait_asynccnt0();
    __syncthreads();

    // A fragment: lane r = M row, elems e<8 -> K=g*8+e ; e>=8 -> K=16+g*8+(e-8)
    Frag16 af[4], bfg[2];
#pragma unroll
    for (int i = 0; i < 4; i++) {
      const __bf16* p = &As[(wm * 64 + i * 16 + r) * LDT];
      af[i].q[0] = *(const u32x4*)(p + g * 8);
      af[i].q[1] = *(const u32x4*)(p + 16 + g * 8);
    }
    // B fragment: lane r = N col, elems e<8 -> K=g*8+e ; e>=8 -> K=16+g*8+(e-8)
#pragma unroll
    for (int j = 0; j < 2; j++) {
      const __bf16* p = &Bs[(wn * 32 + j * 16 + r) * LDT + g * 8];
      bfg[j].q[0] = *(const u32x4*)(p);
      bfg[j].q[1] = *(const u32x4*)(p + 16);
    }
#pragma unroll
    for (int i = 0; i < 4; i++)
#pragma unroll
      for (int j = 0; j < 2; j++) acc[i][j] = wmma_bf16(af[i], bfg[j], acc[i][j]);
    __syncthreads();
  }

  // epilogue: C/D layout -> M = v + 8*g, N = r
  for (int j = 0; j < 2; j++) {
    const int col = n0 + wn * 32 + j * 16 + r;
    const float bv = bias ? bias[col] : 0.f;
    for (int i = 0; i < 4; i++) {
      const int rowb = m0 + wm * 64 + i * 16 + 8 * g;
#pragma unroll
      for (int v = 0; v < 8; v++)
        C[(size_t)(rowb + v) * N + col] = acc[i][j][v] + bv;
    }
  }
}

// ---------------- RoPE + bf16 repack + V transpose + KV-cache scatter ------------
// Also builds vT[b][kvh][d][key] (key contiguous) so attention's V B-operand
// fragments are two contiguous 16B loads instead of 16 strided scalar loads.
__global__ __launch_bounds__(256) void rope_kernel(
    const float* __restrict__ qkv, const int* __restrict__ pos_list,
    const int* __restrict__ slot_mapping,
    __bf16* __restrict__ qb, __bf16* __restrict__ kb, __bf16* __restrict__ vT,
    float* __restrict__ kc, float* __restrict__ vc) {
  const int t = blockIdx.x;
  const int bb = t >> 10;           // t / S
  const int key = t & (S - 1);      // t % S
  const float pos = (float)pos_list[t];
  const int slot = slot_mapping[t];
  const float* row = qkv + (size_t)t * QKVD;

  for (int idx = threadIdx.x; idx < NH * 64; idx += blockDim.x) {
    const int h = idx >> 6, i = idx & 63;
    const float freq = __powf(ROPE_THETA, -(2.f * i) / HD);
    float sv, cv;
    __sincosf(pos * freq, &sv, &cv);
    const float x1 = row[h * HD + i], x2 = row[h * HD + i + 64];
    qb[(size_t)t * NH * HD + h * HD + i]      = (__bf16)(x1 * cv - x2 * sv);
    qb[(size_t)t * NH * HD + h * HD + i + 64] = (__bf16)(x2 * cv + x1 * sv);
  }
  for (int idx = threadIdx.x; idx < NKV * 64; idx += blockDim.x) {
    const int h = idx >> 6, i = idx & 63;
    const float freq = __powf(ROPE_THETA, -(2.f * i) / HD);
    float sv, cv;
    __sincosf(pos * freq, &sv, &cv);
    const float x1 = row[NH * HD + h * HD + i], x2 = row[NH * HD + h * HD + i + 64];
    const float r1 = x1 * cv - x2 * sv, r2 = x2 * cv + x1 * sv;
    kb[(size_t)t * NKV * HD + h * HD + i]      = (__bf16)r1;
    kb[(size_t)t * NKV * HD + h * HD + i + 64] = (__bf16)r2;
    kc[(size_t)slot * NKV * HD + h * HD + i]      = r1;
    kc[(size_t)slot * NKV * HD + h * HD + i + 64] = r2;
  }
  for (int idx = threadIdx.x; idx < NKV * HD; idx += blockDim.x) {
    const int h = idx >> 7, d = idx & 127;
    const float val = row[(NH + NKV) * HD + idx];
    vT[((size_t)(bb * NKV + h) * HD + d) * S + key] = (__bf16)val;
    vc[(size_t)slot * NKV * HD + idx] = val;
  }
}

// ---------------- causal flash attention (one wave per 16-query tile) ----------------
// grid (S/16, NH, B), block 32 (wave32; different q-tiles have different key trip
// counts, so single-wave workgroups keep the barrier trivially safe).
__global__ __launch_bounds__(32) void attn_kernel(
    const __bf16* __restrict__ qb, const __bf16* __restrict__ kb,
    const __bf16* __restrict__ vT, __bf16* __restrict__ ob) {
  __shared__ __bf16 Plds[16 * 32];
  const int lane = threadIdx.x;
  const int g = lane >> 4, r = lane & 15;
  const int qt = blockIdx.x, h = blockIdx.y, b = blockIdx.z;
  const int kvh = h / G;
  const int qp0 = qt * 16;
  const size_t tb = (size_t)b * S;
  const float scale = 0.08838834764831845f;  // HD^-0.5

  Frag16 qf[4];  // Q 16x128 as 4 K-chunks of 32
  {
    const __bf16* qrow = qb + (tb + qp0 + r) * (size_t)(NH * HD) + h * HD;
#pragma unroll
    for (int c = 0; c < 4; c++) {
      qf[c].q[0] = *(const u32x4*)(qrow + c * 32 + g * 8);
      qf[c].q[1] = *(const u32x4*)(qrow + c * 32 + 16 + g * 8);
    }
  }

  f32x8 o[8];
  for (int nt = 0; nt < 8; nt++)
    for (int e = 0; e < 8; e++) o[nt][e] = 0.f;
  float mrow[8], lrow[8];
  for (int v = 0; v < 8; v++) { mrow[v] = -1e30f; lrow[v] = 0.f; }

  const int nblk = qp0 / 32 + 1;  // key blocks of 32 covering keys 0..qp0+15
  for (int jb = 0; jb < nblk; jb++) {
    const int j0 = jb * 32;
    f32x8 s[2];
#pragma unroll
    for (int sub = 0; sub < 2; sub++) {
      for (int e = 0; e < 8; e++) s[sub][e] = 0.f;
      int krow = j0 + sub * 16 + r;
      if (krow >= S) krow = S - 1;  // clamped rows are masked below
      const __bf16* kp = kb + (tb + krow) * (size_t)(NKV * HD) + kvh * HD;
#pragma unroll
      for (int c = 0; c < 4; c++) {
        Frag16 kf;  // B operand: lane r = key (N), elems e<8 -> K(d) = c*32 + g*8 + e, e>=8 -> c*32 + 16 + g*8 + (e-8)
        kf.q[0] = *(const u32x4*)(kp + c * 32 + g * 8);
        kf.q[1] = *(const u32x4*)(kp + c * 32 + 16 + g * 8);
        s[sub] = wmma_bf16(qf[c], kf, s[sub]);
      }
    }
    // scale + causal mask (C layout: key = r, query row = qp0 + 8*g + v)
    float sc[2][8];
#pragma unroll
    for (int sub = 0; sub < 2; sub++) {
      const int key = j0 + sub * 16 + r;
#pragma unroll
      for (int v = 0; v < 8; v++) {
        const int qr = qp0 + 8 * g + v;
        sc[sub][v] = (key <= qr) ? s[sub][v] * scale : -1e30f;
      }
    }
    // online softmax: row reductions over the 16 lanes sharing a half-wave
    float mx[8];
    for (int v = 0; v < 8; v++) mx[v] = fmaxf(sc[0][v], sc[1][v]);
    for (int off = 1; off < 16; off <<= 1)
      for (int v = 0; v < 8; v++) mx[v] = fmaxf(mx[v], __shfl_xor(mx[v], off, 32));
    float alpha[8], mnew[8];
    for (int v = 0; v < 8; v++) {
      mnew[v] = fmaxf(mrow[v], mx[v]);
      alpha[v] = __expf(mrow[v] - mnew[v]);
      mrow[v] = mnew[v];
    }
    float p[2][8], sm[8];
    for (int v = 0; v < 8; v++) sm[v] = 0.f;
    for (int sub = 0; sub < 2; sub++)
      for (int v = 0; v < 8; v++) {
        p[sub][v] = __expf(sc[sub][v] - mnew[v]);
        sm[v] += p[sub][v];
      }
    for (int off = 1; off < 16; off <<= 1)
      for (int v = 0; v < 8; v++) sm[v] += __shfl_xor(sm[v], off, 32);
    for (int v = 0; v < 8; v++) lrow[v] = lrow[v] * alpha[v] + sm[v];
    for (int nt = 0; nt < 8; nt++)
      for (int v = 0; v < 8; v++) o[nt][v] *= alpha[v];

    // P: C layout -> LDS [query row][key], reload in A-operand layout
    for (int sub = 0; sub < 2; sub++)
#pragma unroll
      for (int v = 0; v < 8; v++)
        Plds[(8 * g + v) * 32 + sub * 16 + r] = (__bf16)p[sub][v];
    __syncthreads();
    Frag16 pf;
    pf.q[0] = *(const u32x4*)(&Plds[r * 32 + g * 8]);
    pf.q[1] = *(const u32x4*)(&Plds[r * 32 + 16 + g * 8]);

    // P(16x32) x V(32x128): B operand lane r = d col, elems e<8 -> key = j0+g*8+e, e>=8 -> j0+16+g*8+(e-8).
    // vT has key contiguous -> two b128 loads per fragment (masked P zeroes any
    // beyond-causal keys, so the tail overread rows contribute nothing).
#pragma unroll
    for (int nt = 0; nt < 8; nt++) {
      const int d = nt * 16 + r;
      const __bf16* vrow =
          vT + ((size_t)(b * NKV + kvh) * HD + d) * S + j0 + g * 8;
      Frag16 vf;
      vf.q[0] = *(const u32x4*)(vrow);
      vf.q[1] = *(const u32x4*)(vrow + 16);
      o[nt] = wmma_bf16(pf, vf, o[nt]);
    }
    __syncthreads();
  }

  for (int nt = 0; nt < 8; nt++) {
#pragma unroll
    for (int v = 0; v < 8; v++) {
      const float val = o[nt][v] / lrow[v];
      ob[(tb + qp0 + 8 * g + v) * (size_t)(NH * HD) + h * HD + nt * 16 + r] =
          (__bf16)val;
    }
  }
}

// ---------------- launcher ----------------
extern "C" void kernel_launch(void* const* d_in, const int* in_sizes, int n_in,
                              void* d_out, int out_size, void* d_ws, size_t ws_size,
                              hipStream_t stream) {
  const float* x      = (const float*)d_in[0];
  const float* wqkv   = (const float*)d_in[1];
  const float* bqkv   = (const float*)d_in[2];
  const float* wo     = (const float*)d_in[3];
  const int*   pos    = (const int*)d_in[4];
  const int*   slots  = (const int*)d_in[5];
  // d_in[6], d_in[7] (zero caches) are unused: cache outputs are zero-filled here.

  float* out_f = (float*)d_out;
  float* kc = out_f + (size_t)T * HID;
  float* vc = kc + (size_t)CACHE_ROWS * NKV * HD;

  // workspace carve-up (bf16 copies + fp32 qkv)
  char* ws = (char*)d_ws;
  auto take = [&](size_t bytes) {
    char* p = ws;
    ws += (bytes + 255) & ~(size_t)255;
    return p;
  };
  __bf16* xbf    = (__bf16*)take((size_t)T * HID * 2);
  __bf16* wqkvbf = (__bf16*)take((size_t)HID * QKVD * 2);
  __bf16* wobf   = (__bf16*)take((size_t)(NH * HD) * HID * 2);
  float*  qkvf   = (float*)take((size_t)T * QKVD * 4);
  __bf16* qbf    = (__bf16*)take((size_t)T * NH * HD * 2);
  __bf16* kbf    = (__bf16*)take((size_t)T * NKV * HD * 2);
  __bf16* vTbf   = (__bf16*)take((size_t)Bsz * NKV * HD * S * 2 + 256); // +tail pad
  __bf16* abf    = (__bf16*)take((size_t)T * NH * HD * 2);

  // 1) fp32 -> bf16 operand conversion (bandwidth-bound)
  f32_to_bf16_kernel<<<2048, 256, 0, stream>>>(x, xbf, (size_t)T * HID);
  f32_to_bf16_kernel<<<2048, 256, 0, stream>>>(wqkv, wqkvbf, (size_t)HID * QKVD);
  f32_to_bf16_kernel<<<2048, 256, 0, stream>>>(wo, wobf, (size_t)(NH * HD) * HID);

  // 2) zero the cache output regions (reference caches are all-zero outside slots)
  zero_f32_kernel<<<4096, 256, 0, stream>>>(kc, (size_t)2 * CACHE_ROWS * NKV * HD);

  // 3) QKV projection: [T,HID] x [HID,QKVD] + bias
  {
    dim3 grid(QKVD / 128, T / 128);
    gemm_bf16_wmma<<<grid, 256, 0, stream>>>(xbf, wqkvbf, bqkv, qkvf, T, QKVD, HID);
  }

  // 4) RoPE, bf16 repack, V transpose, KV cache scatter
  rope_kernel<<<T, 256, 0, stream>>>(qkvf, pos, slots, qbf, kbf, vTbf, kc, vc);

  // 5) causal flash attention
  {
    dim3 grid(S / 16, NH, Bsz);
    attn_kernel<<<grid, 32, 0, stream>>>(qbf, kbf, vTbf, abf);
  }

  // 6) output projection: [T,4096] x [4096,HID] -> d_out
  {
    dim3 grid(HID / 128, T / 128);
    gemm_bf16_wmma<<<grid, 256, 0, stream>>>(abf, wobf, nullptr, out_f, T, HID,
                                             NH * HD);
  }
}
